// PNSNet_7653631721516
// MI455X (gfx1250) — matrix operand order repaired and checked by reading, not compiled, not run on hardware
//
#include <hip/hip_runtime.h>
#include <hip/hip_bf16.h>

typedef __attribute__((ext_vector_type(16))) _Float16 v16h;
typedef __attribute__((ext_vector_type(8)))  _Float16 v8h;
typedef __attribute__((ext_vector_type(8)))  float    v8f;

#define NB 48
#define HW 16384            // 128*128

#if defined(__has_builtin)
#  if __has_builtin(__builtin_amdgcn_global_load_async_to_lds_b128)
#    define USE_ASYNC_LDS 1
#  endif
#endif
#ifndef USE_ASYNC_LDS
#  define USE_ASYNC_LDS 0
#endif

// 16-byte int vector type matching the async-to-LDS builtin's pointee type
typedef __attribute__((__vector_size__(4 * sizeof(int)))) int i4;
typedef __attribute__((address_space(1))) i4 gi4;   // global
typedef __attribute__((address_space(3))) i4 li4;   // LDS

__device__ __forceinline__ float mishf(float x){
  float sp = (x > 20.f) ? x : log1pf(expf(x));
  return x * tanhf(sp);
}
__device__ __forceinline__ int imin(int a,int b){ return a<b?a:b; }
__device__ __forceinline__ int imax(int a,int b){ return a>b?a:b; }

// ---------------------------------------------------------------------------
// K0: zero the stats region (graph replays must be deterministic)
// ---------------------------------------------------------------------------
__global__ void k_zero(float* __restrict__ p, int n){
  int i = blockIdx.x*blockDim.x + threadIdx.x;
  if (i < n) p[i] = 0.f;
}

// ---------------------------------------------------------------------------
// K1: in0 = conv1x1(high_fea[48,32,128,128], W_c0[16,32]) -> f16 [48,16,HW]
// GEMM M=16 pixels, N=16 cout, K=32 cin, one v_wmma_f32_16x16x32_f16 per tile
// ---------------------------------------------------------------------------
__global__ __launch_bounds__(256) void k_in0(const float* __restrict__ x,
                                             const float* __restrict__ w,
                                             _Float16* __restrict__ out){
  const int lane = threadIdx.x & 31, wave = threadIdx.x >> 5;
  const int b = blockIdx.x >> 5, blk = blockIdx.x & 31;
  const int grp = lane >> 4, n = lane & 15;
  v16h bm;
  #pragma unroll
  for (int s = 0; s < 16; ++s) bm[s] = (_Float16)w[n*32 + (grp<<4) + s];
  const float* xb = x + (size_t)b*32*HW;
  _Float16* ob = out + (size_t)b*16*HW;
  #pragma unroll
  for (int t = 0; t < 4; ++t){
    const int pix = (blk<<9) + (wave<<6) + (t<<4);
    v16h am;
    #pragma unroll
    for (int s = 0; s < 16; ++s){
      int k = (grp<<4) + s;                       // A K-slot mapping (contiguous, matches B)
      am[s] = (_Float16)xb[(size_t)k*HW + pix + n];
    }
    v8f c = {};
    c = __builtin_amdgcn_wmma_f32_16x16x32_f16(false, am, false, bm, (short)0, c, false, false);
    v8h o;
    #pragma unroll
    for (int i = 0; i < 8; ++i) o[i] = (_Float16)c[i];
    *(v8h*)(ob + (size_t)n*HW + pix + (grp<<3)) = o;
  }
}

// ---------------------------------------------------------------------------
// K2: low = conv1x1(low_fea[48,24,...], W_low[16,24]) -> f32, + GN stats (2 grp)
// ---------------------------------------------------------------------------
__global__ __launch_bounds__(256) void k_low(const float* __restrict__ x,
                                             const float* __restrict__ w,
                                             float* __restrict__ out,
                                             float* __restrict__ stats){
  __shared__ float sred[4];
  const int lane = threadIdx.x & 31, wave = threadIdx.x >> 5;
  const int b = blockIdx.x >> 5, blk = blockIdx.x & 31;
  const int grp = lane >> 4, n = lane & 15;
  v16h bm;
  #pragma unroll
  for (int s = 0; s < 16; ++s){
    int k = (grp<<4) + s;
    bm[s] = (_Float16)((k < 24) ? w[n*24 + k] : 0.f);
  }
  const float* xb = x + (size_t)b*24*HW;
  float* ob = out + (size_t)b*16*HW;
  float s1 = 0.f, s2 = 0.f;
  #pragma unroll
  for (int t = 0; t < 4; ++t){
    const int pix = (blk<<9) + (wave<<6) + (t<<4);
    v16h am;
    #pragma unroll
    for (int s = 0; s < 16; ++s){
      int k = (grp<<4) + s;
      am[s] = (k < 24) ? (_Float16)xb[(size_t)k*HW + pix + n] : (_Float16)0.f;
    }
    v8f c = {};
    c = __builtin_amdgcn_wmma_f32_16x16x32_f16(false, am, false, bm, (short)0, c, false, false);
    *(v8f*)(ob + (size_t)n*HW + pix + (grp<<3)) = c;
    #pragma unroll
    for (int i = 0; i < 8; ++i){ s1 += c[i]; s2 += c[i]*c[i]; }
  }
  if (threadIdx.x < 4) sred[threadIdx.x] = 0.f;
  __syncthreads();
  const int g = n >> 3;
  atomicAdd(&sred[g*2+0], s1);
  atomicAdd(&sred[g*2+1], s2);
  __syncthreads();
  if (threadIdx.x < 4) atomicAdd(&stats[b*4 + threadIdx.x], sred[threadIdx.x]);
}

// ---------------------------------------------------------------------------
// K3/K6: 3x3 convs as implicit GEMM, K = tap*16 + cin (144 -> 5 WMMAs of K=32).
// Input tile (16ch x 8 rows x 128) staged to LDS via async-to-LDS copies; tap
// gathers then come from LDS (halo rows pre-zeroed, so only x needs checking).
// MODE 0: b1/b2 fused (block-diag weights, dil 1 for cin<8, 2 for cin>=8), 1 GN group.
// MODE 1: W_ref 16->16 dil 1, 2 GN groups.
// ---------------------------------------------------------------------------
template<int MODE>
__global__ __launch_bounds__(256) void k_conv3(const _Float16* __restrict__ in,
                                               const float* __restrict__ w1,
                                               const float* __restrict__ w2,
                                               float* __restrict__ out,
                                               float* __restrict__ stats){
  __shared__ _Float16 sh[16*8*128];   // 32 KB: [ch][row(8)][x(128)]
  __shared__ float sred[4];
  const int lane = threadIdx.x & 31, wave = threadIdx.x >> 5;
  const int b = blockIdx.x >> 5, blk = blockIdx.x & 31;
  const int grp = lane >> 4, n = lane & 15;
  const int y0 = blk << 2;            // first of this block's 4 output rows

  // ---- build B (weights), K = tap*16 + cin, 5 chunks of 32 ----
  v16h bm[5];
  #pragma unroll
  for (int q = 0; q < 5; ++q){
    #pragma unroll
    for (int s = 0; s < 16; ++s){
      int kk = q*32 + (grp<<4) + s;
      float v = 0.f;
      if (kk < 144){
        int cin = kk & 15, tap = kk >> 4;
        if (MODE == 1) v = w1[(n*16 + cin)*9 + tap];
        else {
          if (n < 8 && cin < 8)        v = w1[(n*8 + cin)*9 + tap];
          else if (n >= 8 && cin >= 8) v = w2[((n-8)*8 + (cin-8))*9 + tap];
        }
      }
      bm[q][s] = (_Float16)v;
    }
  }

  // ---- stage input tile to LDS: 2048 x 16B chunks, 8 per thread ----
  const _Float16* ib = in + (size_t)b*16*HW;
  #pragma unroll
  for (int j = 0; j < 8; ++j){
    int cidx = (int)threadIdx.x + (j << 8);       // 0..2047
    int ch  = cidx >> 7;
    int row = (cidx >> 4) & 7;
    int xc  = cidx & 15;
    int yy  = y0 - 2 + row;
    _Float16* ldst = &sh[(((ch<<3) + row) << 7) + (xc<<3)];
    if (yy >= 0 && yy < 128){
      const _Float16* gsrc = ib + (size_t)ch*HW + (yy<<7) + (xc<<3);
#if USE_ASYNC_LDS
      __builtin_amdgcn_global_load_async_to_lds_b128((gi4*)gsrc, (li4*)ldst, 0, 0);
#else
      *(v8h*)ldst = *(const v8h*)gsrc;
#endif
    } else {
      v8h z = {};
      *(v8h*)ldst = z;                            // zero-fill halo rows
    }
  }
#if USE_ASYNC_LDS
#  if __has_builtin(__builtin_amdgcn_s_wait_asynccnt)
  __builtin_amdgcn_s_wait_asynccnt(0);
#  else
  asm volatile("s_wait_asynccnt 0" ::: "memory");
#  endif
#endif
  __syncthreads();

  // ---- implicit GEMM from LDS ----
  float* ob = out + (size_t)b*16*HW;
  float s1 = 0.f, s2 = 0.f;
  #pragma unroll
  for (int t = 0; t < 4; ++t){
    const int pix = (blk<<9) + (wave<<6) + (t<<4);
    const int p = pix + n;
    const int yl = (p >> 7) - y0 + 2;             // LDS row of this output pixel
    const int x = p & 127;
    v8f c = {};
    #pragma unroll
    for (int q = 0; q < 5; ++q){
      v16h am;
      #pragma unroll
      for (int s = 0; s < 16; ++s){
        int kk = q*32 + (grp<<4) + s;
        _Float16 v = (_Float16)0.f;
        if (kk < 144){
          int cin = kk & 15, tap = kk >> 4;
          int d = (MODE == 1) ? 1 : ((cin < 8) ? 1 : 2);
          int dy = (tap/3 - 1)*d, dx = (tap%3 - 1)*d;
          int xx = x + dx;
          if (xx >= 0 && xx < 128)
            v = sh[(((cin<<3) + yl + dy) << 7) + xx];
        }
        am[s] = v;
      }
      c = __builtin_amdgcn_wmma_f32_16x16x32_f16(false, am, false, bm[q], (short)0, c, false, false);
    }
    *(v8f*)(ob + (size_t)n*HW + pix + (grp<<3)) = c;
    #pragma unroll
    for (int i = 0; i < 8; ++i){ s1 += c[i]; s2 += c[i]*c[i]; }
  }

  // ---- GN statistics ----
  const int ngroups = (MODE == 1) ? 2 : 1;
  if (threadIdx.x < 2*ngroups) sred[threadIdx.x] = 0.f;
  __syncthreads();
  const int g = (MODE == 1) ? (n >> 3) : 0;
  atomicAdd(&sred[g*2+0], s1);
  atomicAdd(&sred[g*2+1], s2);
  __syncthreads();
  if (threadIdx.x < 2*ngroups)
    atomicAdd(&stats[b*2*ngroups + threadIdx.x], sred[threadIdx.x]);
}

// ---------------------------------------------------------------------------
// Fold GN stats + gamma/beta into per-(batch,channel) scale/bias
// ---------------------------------------------------------------------------
__global__ void k_fin(const float* __restrict__ stats, const float* __restrict__ gamma,
                      const float* __restrict__ beta, int groups, float* __restrict__ sb){
  int i = blockIdx.x*blockDim.x + threadIdx.x;
  if (i >= NB*16) return;
  int b = i >> 4, cch = i & 15;
  int cpg = 16 / groups;
  int g = cch / cpg;
  float cnt = (float)cpg * (float)HW;
  float su = stats[(b*groups + g)*2 + 0];
  float sq = stats[(b*groups + g)*2 + 1];
  float mu = su/cnt, var = sq/cnt - mu*mu;
  float rs = rsqrtf(var + 1e-5f);
  float sc = rs * gamma[cch];
  sb[2*i+0] = sc;
  sb[2*i+1] = beta[cch] - mu*sc;
}

// ---------------------------------------------------------------------------
// K5: r = mish(GN(high) + GN(low)) -> f16
// ---------------------------------------------------------------------------
__global__ __launch_bounds__(256) void k_mix(const float* __restrict__ hi,
                                             const float* __restrict__ lo,
                                             const float* __restrict__ sbH,
                                             const float* __restrict__ sbL,
                                             _Float16* __restrict__ r){
  size_t i = (size_t)blockIdx.x*blockDim.x + threadIdx.x;   // NB*16*HW
  int bc = (int)(i >> 14);
  float x = hi[i]*sbH[2*bc] + sbH[2*bc+1] + lo[i]*sbL[2*bc] + sbL[2*bc+1];
  r[i] = (_Float16)mishf(x);
}

// ---------------------------------------------------------------------------
// K8: seg = conv1x1(mish(GN(ref)), W_seg[1,16]) -> f32 [48,HW]
// ---------------------------------------------------------------------------
__global__ __launch_bounds__(256) void k_segk(const float* __restrict__ ref,
                                              const float* __restrict__ sbR,
                                              const float* __restrict__ wseg,
                                              float* __restrict__ seg){
  int i = blockIdx.x*blockDim.x + threadIdx.x;              // NB*HW
  int b = i >> 14, pix = i & (HW-1);
  float acc = 0.f;
  #pragma unroll
  for (int cch = 0; cch < 16; ++cch){
    int bc = b*16 + cch;
    float x = ref[(size_t)bc*HW + pix]*sbR[2*bc] + sbR[2*bc+1];
    acc += wseg[cch] * mishf(x);
  }
  seg[i] = acc;
}

// ---------------------------------------------------------------------------
// K9: 4x bilinear upsample (half-pixel centers, edge clamp) + sigmoid
// ---------------------------------------------------------------------------
__global__ __launch_bounds__(256) void k_up(const float* __restrict__ seg,
                                            float* __restrict__ out){
  size_t i = (size_t)blockIdx.x*blockDim.x + threadIdx.x;   // NB*512*512
  int b = (int)(i >> 18);
  int rem = (int)(i & 262143);
  int oy = rem >> 9, ox = rem & 511;
  float fy = oy*0.25f - 0.375f, fx = ox*0.25f - 0.375f;
  float y0f = floorf(fy), x0f = floorf(fx);
  float wy = fy - y0f, wx = fx - x0f;
  int y0 = (int)y0f, x0 = (int)x0f;
  int y0c = imin(imax(y0,   0), 127), y1c = imin(imax(y0+1, 0), 127);
  int x0c = imin(imax(x0,   0), 127), x1c = imin(imax(x0+1, 0), 127);
  const float* sp = seg + (size_t)b*HW;
  float v00 = sp[(y0c<<7)+x0c], v01 = sp[(y0c<<7)+x1c];
  float v10 = sp[(y1c<<7)+x0c], v11 = sp[(y1c<<7)+x1c];
  float v = v00*(1.f-wy)*(1.f-wx) + v01*(1.f-wy)*wx
          + v10*wy*(1.f-wx)       + v11*wy*wx;
  out[i] = 1.f/(1.f + expf(-v));
}

// ---------------------------------------------------------------------------
extern "C" void kernel_launch(void* const* d_in, const int* in_sizes, int n_in,
                              void* d_out, int out_size, void* d_ws, size_t ws_size,
                              hipStream_t stream){
  const float* low_fea  = (const float*)d_in[0];
  const float* high_fea = (const float*)d_in[1];
  const float* W_c0  = (const float*)d_in[2];
  const float* W_c1  = (const float*)d_in[3];
  const float* W_c2  = (const float*)d_in[4];
  const float* g_bn  = (const float*)d_in[5];
  const float* b_bn  = (const float*)d_in[6];
  const float* W_low = (const float*)d_in[7];
  const float* g_low = (const float*)d_in[8];
  const float* b_low = (const float*)d_in[9];
  const float* W_ref = (const float*)d_in[10];
  const float* g_ref = (const float*)d_in[11];
  const float* b_ref = (const float*)d_in[12];
  const float* W_seg = (const float*)d_in[13];
  float* out = (float*)d_out;

  char* ws = (char*)d_ws;
  _Float16* bufA = (_Float16*)(ws);               // 25,165,824 B : in0, later r
  float*    bufB = (float*)(ws + 25165824);       // 50,331,648 B : high, later ref
  float*    bufC = (float*)(ws + 75497472);       // 50,331,648 B : low,  later seg
  float*    stats= (float*)(ws + 125829120);      // 480 floats
  float*    stH  = stats;                         // [48][1][2]
  float*    stL  = stats + 96;                    // [48][2][2]
  float*    stR  = stats + 288;                   // [48][2][2]
  float*    sb   = (float*)(ws + 125831040);      // 3 * 48*16*2 floats
  float*    sbH  = sb;
  float*    sbL  = sb + 1536;
  float*    sbR  = sb + 3072;

  k_zero<<<2, 256, 0, stream>>>(stats, 480);
  k_in0 <<<NB*32, 256, 0, stream>>>(high_fea, W_c0, bufA);
  k_low <<<NB*32, 256, 0, stream>>>(low_fea, W_low, bufC, stL);
  k_conv3<0><<<NB*32, 256, 0, stream>>>(bufA, W_c1, W_c2, bufB, stH);
  k_fin <<<3, 256, 0, stream>>>(stH, g_bn,  b_bn,  1, sbH);
  k_fin <<<3, 256, 0, stream>>>(stL, g_low, b_low, 2, sbL);
  k_mix <<<49152, 256, 0, stream>>>(bufB, bufC, sbH, sbL, bufA);
  k_conv3<1><<<NB*32, 256, 0, stream>>>(bufA, W_ref, W_ref, bufB, stR);
  k_fin <<<3, 256, 0, stream>>>(stR, g_ref, b_ref, 2, sbR);
  k_segk<<<3072, 256, 0, stream>>>(bufB, sbR, W_seg, bufC);
  k_up  <<<49152, 256, 0, stream>>>(bufC, out);
}
